// ADM_6511170421537
// MI455X (gfx1250) — matrix operand order repaired and checked by reading, not compiled, not run on hardware
//
#include <hip/hip_runtime.h>
#include <hip/hip_bf16.h>

// ---------------------------------------------------------------------------
// Fused autoregressive MLP for MI455X (gfx1250, wave32).
//   trunk: 64->256->256->256 (ReLU), then 8 AR steps:
//     x = relu(h @ W_in[i][:256+i]); x = relu(x @ W_h[i]); out = relu(x @ W_out[i])
//     mean/log_std -> sample/logp; h = concat(h, mean)
// Strategy: keep h/x tiles (128 rows) entirely in LDS (3 buffers, 222 KB of the
// 320 KB WGP LDS), run all 256-wide GEMMs with v_wmma_f32_16x16x32_f16 (f16 in,
// f32 accumulate), handle the ragged K=256+i via zero-padded K=288.
// Weights are pre-packed once per launch into the WMMA B-fragment lane layout.
// ---------------------------------------------------------------------------

typedef _Float16 v16h __attribute__((ext_vector_type(16)));
typedef float    v8f  __attribute__((ext_vector_type(8)));
typedef unsigned short u16;
typedef unsigned int   u32;

#define BATCH_TOTAL 65536
#define BM   128          // rows per workgroup
#define SH   296          // LDS row stride in halves (148 dwords == 20 mod 64 -> conflict-free b128)
#define NWAVE 8

#define LOG_STD_MIN (-20.0f)
#define LOG_STD_MAX (2.0f)
#define LOG_2PI     (1.8378770664093453f)

// packed-weight workspace offsets (in u32 units); each tile = 32 lanes * 8 dwords
#define OFF_SW0 0            // K=64  : 2*16 tiles  -> 8192
#define OFF_SW1 8192         // K=256 : 8*16 tiles  -> 32768
#define OFF_SW2 40960        // K=256 : 8*16 tiles  -> 32768
#define OFF_WIN 73728        // 8 mats, K=288 pad : 9*16 tiles each -> 36864 per mat
#define WIN_STRIDE 36864
#define OFF_WH  368640       // 8 mats, K=256 : 8*16 tiles each -> 32768 per mat
#define WH_STRIDE 32768
#define N_TILES_TOTAL (32 + 128 + 128 + 8*144 + 8*128)   // 2464

union FragU { uint4 u4[2]; v16h h; };

__device__ __forceinline__ u32 pack2h(float a, float b) {
  union { _Float16 h[2]; u32 u; } x;
  x.h[0] = (_Float16)a; x.h[1] = (_Float16)b;
  return x.u;
}

// ---------------------------------------------------------------------------
// Weight packing: f32 (K x 256, row-major) -> per-lane WMMA B fragments (f16).
// Fragment layout per (ktile, ntile): lane L<16 holds col n0+L, K rows k0+0..15
// (pairs per dword); lane L>=16 holds col n0+L-16, K rows k0+16..31.
// ---------------------------------------------------------------------------
__global__ void ADM_pack_weights(const float* __restrict__ sW0,
                                 const float* __restrict__ sW1,
                                 const float* __restrict__ sW2,
                                 const float* __restrict__ W_in,
                                 const float* __restrict__ W_h,
                                 u32* __restrict__ wp) {
  int gid = blockIdx.x * blockDim.x + threadIdx.x;
  if (gid >= N_TILES_TOTAL * 32) return;
  int tile = gid >> 5;
  int ln   = gid & 31;

  const float* src; int Ksrc; int kt, nt; u32* dst;
  if (tile < 32) {                       // sW0: 64 x 256
    int lt = tile;      src = sW0; Ksrc = 64;
    kt = lt >> 4; nt = lt & 15;
    dst = wp + OFF_SW0 + (((kt * 16 + nt) * 32) + ln) * 8;
  } else if (tile < 160) {               // sW1: 256 x 256
    int lt = tile - 32; src = sW1; Ksrc = 256;
    kt = lt >> 4; nt = lt & 15;
    dst = wp + OFF_SW1 + (((kt * 16 + nt) * 32) + ln) * 8;
  } else if (tile < 288) {               // sW2: 256 x 256
    int lt = tile - 160; src = sW2; Ksrc = 256;
    kt = lt >> 4; nt = lt & 15;
    dst = wp + OFF_SW2 + (((kt * 16 + nt) * 32) + ln) * 8;
  } else if (tile < 288 + 8 * 144) {     // W_in[i]: 263 x 256, pad K->288
    int lt = tile - 288; int i = lt / 144; lt -= i * 144;
    src = W_in + (long)i * 263 * 256; Ksrc = 263;
    kt = lt >> 4; nt = lt & 15;
    dst = wp + OFF_WIN + i * WIN_STRIDE + (((kt * 16 + nt) * 32) + ln) * 8;
  } else {                               // W_h[i]: 256 x 256
    int lt = tile - (288 + 8 * 144); int i = lt / 128; lt -= i * 128;
    src = W_h + (long)i * 256 * 256; Ksrc = 256;
    kt = lt >> 4; nt = lt & 15;
    dst = wp + OFF_WH + i * WH_STRIDE + (((kt * 16 + nt) * 32) + ln) * 8;
  }

  int col   = nt * 16 + (ln & 15);
  int kbase = kt * 32 + ((ln >> 4) & 1) * 16;
#pragma unroll
  for (int v = 0; v < 8; v++) {
    int k0 = kbase + 2 * v, k1 = k0 + 1;
    float a = (k0 < Ksrc) ? src[(long)k0 * 256 + col] : 0.0f;
    float b = (k1 < Ksrc) ? src[(long)k1 * 256 + col] : 0.0f;
    dst[v] = pack2h(a, b);
  }
}

// ---------------------------------------------------------------------------
// One 128 x 256 GEMM layer (K = KT*32) from LDS f16 activations, WMMA f32 acc,
// bias + ReLU epilogue, f16 result back to LDS. Each wave owns one 16-row tile.
// ---------------------------------------------------------------------------
template <int KT>
__device__ __forceinline__ void gemm_layer(const u16* __restrict__ lin,
                                           u16* __restrict__ lout,
                                           const u32* __restrict__ wp,
                                           const float* __restrict__ bias,
                                           int wv, int ln) {
  const int r0      = wv * 16;
  const int laneCol = ln & 15;
  const int hiHalf  = (ln >> 4) & 1;

  // A fragments for this wave's 16 rows, all K tiles, held in VGPRs.
  FragU A[KT];
#pragma unroll
  for (int kt = 0; kt < KT; kt++) {
    const u16* ap = lin + (r0 + laneCol) * SH + kt * 32 + hiHalf * 16;
    A[kt].u4[0] = *(const uint4*)(ap);
    A[kt].u4[1] = *(const uint4*)(ap + 8);
  }

#pragma unroll
  for (int nt = 0; nt < 16; nt++) {
    v8f acc = {0.f, 0.f, 0.f, 0.f, 0.f, 0.f, 0.f, 0.f};
#pragma unroll
    for (int kt = 0; kt < KT; kt++) {
      FragU B;
      const u32* bp = wp + (((kt * 16 + nt) * 32) + ln) * 8;
      B.u4[0] = *(const uint4*)(bp);
      B.u4[1] = *(const uint4*)(bp + 4);
      acc = __builtin_amdgcn_wmma_f32_16x16x32_f16(
          false, A[kt].h, false, B.h, (short)0, acc, false, false);
    }
    const int col = nt * 16 + laneCol;
    const float bv = bias[col];
    u16* op = lout + (r0 + hiHalf * 8) * SH + col;
#pragma unroll
    for (int v = 0; v < 8; v++) {
      float x = acc[v] + bv;
      x = fmaxf(x, 0.0f);
      _Float16 hx = (_Float16)x;
      op[v * SH] = *(u16*)&hx;
    }
  }
  __syncthreads();
}

// ---------------------------------------------------------------------------
// Fused network kernel: 512 blocks x 256 threads (8 waves), 128 rows per block.
// ---------------------------------------------------------------------------
__global__ __launch_bounds__(256, 1)
void ADM_fused(const float* __restrict__ inputs, const float* __restrict__ eps,
               const u32* __restrict__ wp,
               const float* __restrict__ sb0, const float* __restrict__ sb1,
               const float* __restrict__ sb2,
               const float* __restrict__ b_in, const float* __restrict__ b_h,
               const float* __restrict__ W_out, const float* __restrict__ b_out,
               float* __restrict__ out_mean, float* __restrict__ out_sample,
               float* __restrict__ out_logp) {
  __shared__ __align__(16) u16 lds[3 * BM * SH];   // 227,328 B of 320 KB WGP LDS
  u16* H  = lds;
  u16* X1 = lds + BM * SH;
  u16* X2 = lds + 2 * BM * SH;

  const int t  = threadIdx.x;
  const int wv = t >> 5;
  const int ln = t & 31;
  const long g0 = (long)blockIdx.x * BM;

  // ---- stage inputs (f32 -> f16) into X1; zero H pad columns 256..287 ----
  {
    const int r     = t >> 1;
    const int cbase = (t & 1) * 32;
    const float* src = inputs + (g0 + r) * 64 + cbase;
    u16* dst = X1 + r * SH + cbase;
#pragma unroll
    for (int j = 0; j < 32; j += 2) {
      *(u32*)(dst + j) = pack2h(src[j], src[j + 1]);
    }
    u32* hz = (u32*)(H + r * SH + 256) + (t & 1) * 8;
#pragma unroll
    for (int j = 0; j < 8; j++) hz[j] = 0u;
  }
  __syncthreads();

  // ---- trunk ----
  gemm_layer<2>(X1, X2, wp + OFF_SW0, sb0, wv, ln);   // 64  -> 256
  gemm_layer<8>(X2, X1, wp + OFF_SW1, sb1, wv, ln);   // 256 -> 256
  gemm_layer<8>(X1, H,  wp + OFF_SW2, sb2, wv, ln);   // 256 -> 256 (latent h)

  // ---- autoregressive steps ----
  for (int i = 0; i < 8; i++) {
    gemm_layer<9>(H,  X1, wp + OFF_WIN + i * WIN_STRIDE, b_in + i * 256, wv, ln);
    gemm_layer<8>(X1, X2, wp + OFF_WH  + i * WH_STRIDE,  b_h  + i * 256, wv, ln);

    // final 256 -> 2 layer in fp32 VALU: thread pair (c=0 mean, c=1 log_std)
    const int r = t >> 1;
    const int c = t & 1;
    const _Float16* xrow = (const _Float16*)(X2 + r * SH);
    const float* wo = W_out + (long)i * 512 + c;
    float acc = b_out[i * 2 + c];
#pragma unroll 8
    for (int k = 0; k < 256; k++) acc += (float)xrow[k] * wo[k * 2];
    acc = fmaxf(acc, 0.0f);                 // relu on both outputs
    float other = __shfl_xor(acc, 1, 32);   // wave32 pair exchange

    if (c == 0) {
      const float mean    = acc;
      const float log_std = fminf(fmaxf(other, LOG_STD_MIN), LOG_STD_MAX);
      const float std     = expf(log_std);
      const long  row     = g0 + r;
      const float e       = eps[row * 8 + i];
      const float sample  = mean + std * e;
      const float z       = (sample - mean) / std;
      const float logp    = -0.5f * z * z - log_std - 0.5f * LOG_2PI;
      out_mean[row * 8 + i]   = mean;
      out_sample[row * 8 + i] = sample;
      out_logp[row * 8 + i]   = logp;
      _Float16 hm = (_Float16)mean;         // append mean to h
      H[r * SH + 256 + i] = *(u16*)&hm;
    }
    __syncthreads();
  }
}

extern "C" void kernel_launch(void* const* d_in, const int* in_sizes, int n_in,
                              void* d_out, int out_size, void* d_ws, size_t ws_size,
                              hipStream_t stream) {
  const float* inputs = (const float*)d_in[0];
  const float* eps    = (const float*)d_in[1];
  const float* sW0    = (const float*)d_in[2];
  const float* sb0    = (const float*)d_in[3];
  const float* sW1    = (const float*)d_in[4];
  const float* sb1    = (const float*)d_in[5];
  const float* sW2    = (const float*)d_in[6];
  const float* sb2    = (const float*)d_in[7];
  const float* W_in   = (const float*)d_in[8];
  const float* b_in   = (const float*)d_in[9];
  const float* W_h    = (const float*)d_in[10];
  const float* b_h    = (const float*)d_in[11];
  const float* W_out  = (const float*)d_in[12];
  const float* b_out  = (const float*)d_in[13];

  float* out = (float*)d_out;
  u32*   wp  = (u32*)d_ws;   // needs ~2.41 MB of scratch

  // 1) pack all weights into WMMA B-fragment layout (f16, zero-padded K)
  const int pack_threads = N_TILES_TOTAL * 32;
  ADM_pack_weights<<<(pack_threads + 255) / 256, 256, 0, stream>>>(
      sW0, sW1, sW2, W_in, W_h, wp);

  // 2) fused network: 65536 rows / 128 per block
  ADM_fused<<<BATCH_TOTAL / BM, 256, 0, stream>>>(
      inputs, eps, wp, sb0, sb1, sb2, b_in, b_h, W_out, b_out,
      out, out + (long)BATCH_TOTAL * 8, out + 2L * BATCH_TOTAL * 8);
}
